// BuildingGNN_87067577024957
// MI455X (gfx1250) — matrix operand order repaired and checked by reading, not compiled, not run on hardware
//
#include <hip/hip_runtime.h>
#include <hip/hip_bf16.h>
#include <math.h>

typedef __attribute__((ext_vector_type(16))) _Float16 v16h;
typedef __attribute__((ext_vector_type(8)))  _Float16 v8h;
typedef __attribute__((ext_vector_type(8)))  float    v8f;

// ---------------------------------------------------------------------------
// Degree / normalization
// ---------------------------------------------------------------------------
__global__ void init_deg_kernel(float* deg, int n) {
    int i = blockIdx.x * blockDim.x + threadIdx.x;
    if (i < n) deg[i] = 1.0f;                 // self loop contributes 1
}

__global__ void count_deg_kernel(const int* __restrict__ ei, int E, float* deg) {
    int e = blockIdx.x * blockDim.x + threadIdx.x;
    if (e < E) atomicAdd(&deg[ei[E + e]], 1.0f);   // dst row of edge_index
}

__global__ void finalize_dinv_kernel(float* deg, int n) {
    int i = blockIdx.x * blockDim.x + threadIdx.x;
    if (i < n) deg[i] = rsqrtf(deg[i]);       // deg >= 1 always (self loops)
}

// ---------------------------------------------------------------------------
// Pack activations fp32 -> f16 row-major (optional fused ReLU).
// 8 elements per thread: 2x float4 in, 1x v8h (b128) out.
// ---------------------------------------------------------------------------
__global__ void pack_a_kernel(const float* __restrict__ A,
                              _Float16* __restrict__ out,
                              int total8, int relu) {
    int t = blockIdx.x * blockDim.x + threadIdx.x;
    if (t >= total8) return;
    const float4* p = reinterpret_cast<const float4*>(A) + (size_t)t * 2;
    float4 v0 = p[0], v1 = p[1];
    float f[8] = {v0.x, v0.y, v0.z, v0.w, v1.x, v1.y, v1.z, v1.w};
    v8h h;
    #pragma unroll
    for (int j = 0; j < 8; ++j) {
        float v = f[j];
        if (relu) v = fmaxf(v, 0.0f);
        h[j] = (_Float16)v;
    }
    *(reinterpret_cast<v8h*>(out) + t) = h;
}

// ---------------------------------------------------------------------------
// Pack weights fp32 [K,N] -> f16 WMMA B fragments.
// Fragment order: frag = tileN * (K/32) + kblk ; within fragment 512 halfs:
//   element [lane*16 + j] = W[(kblk*32 + 16*(lane>>4) + j) * N + tileN*16 + (lane&15)]
// (Contiguous k order chosen here: lane = n + 16*hi, half j -> K = j + 16*hi.)
// ---------------------------------------------------------------------------
__global__ void pack_b_kernel(const float* __restrict__ W,
                              _Float16* __restrict__ out, int K, int N) {
    int idx = blockIdx.x * blockDim.x + threadIdx.x;
    if (idx >= K * N) return;
    int frag = idx >> 9;          // / 512
    int r    = idx & 511;
    int lane = r >> 4;
    int j    = r & 15;
    int kblocks = K >> 5;
    int tileN = frag / kblocks;
    int kblk  = frag - tileN * kblocks;
    int hi  = lane >> 4;
    int col = tileN * 16 + (lane & 15);
    int krow = kblk * 32 + 16 * hi + j;
    out[idx] = (_Float16)W[(size_t)krow * N + col];
}

// ---------------------------------------------------------------------------
// WMMA GEMM on packed operands: C[M,N] = Ah[M,K] @ B + (bias).
// One wave computes a 16x(16*NT) output strip: one A fragment feeds NT
// v_wmma_f32_16x16x32_f16 ops per K-step. K templated -> fully unrolled.
//
// A 16x32 f16 k order used here (contiguous): lane = l + 16*hi (l = row M);
//   halfs 0..15 -> K = k0 + 16*hi .. +15 (matches packed B order).
// C/D 16x16 f32: vgpr v, lane = l + 16*hi -> row M = v + 8*hi, col N = l.
// M must be a multiple of 16 (50000 = 3125*16).
// ---------------------------------------------------------------------------
template <int NT, int K>
__global__ void wmma_gemm_packed_kernel(const _Float16* __restrict__ Ah,
                                        const _Float16* __restrict__ Bp,
                                        const float* __restrict__ bias,
                                        float* __restrict__ C,
                                        int M, int N, int add_bias) {
    const int wid  = (blockIdx.x * blockDim.x + threadIdx.x) >> 5;
    const int lane = threadIdx.x & 31;
    const int tilesN = N >> 4;
    const int ncol   = tilesN / NT;
    const int tilesM = M >> 4;
    if (wid >= tilesM * ncol) return;

    const int tileM  = wid / ncol;
    const int tileN0 = (wid % ncol) * NT;
    const int l  = lane & 15;
    const int hi = lane >> 4;
    constexpr int KB = K >> 5;

    const _Float16* __restrict__ arow = Ah + (size_t)(tileM * 16 + l) * K;

    v8f acc[NT];
    #pragma unroll
    for (int t = 0; t < NT; ++t) acc[t] = (v8f){};

    #pragma unroll
    for (int kblk = 0; kblk < KB; ++kblk) {
        const int k0 = kblk * 32;
        // A fragment: two 16B vector loads of f16.
        v8h a_lo = *reinterpret_cast<const v8h*>(arow + k0 + 16 * hi);
        v8h a_hi = *reinterpret_cast<const v8h*>(arow + k0 + 16 * hi + 8);
        v16h a = __builtin_shufflevector(a_lo, a_hi, 0, 1, 2, 3, 4, 5, 6, 7,
                                         8, 9, 10, 11, 12, 13, 14, 15);
        #pragma unroll
        for (int t = 0; t < NT; ++t) {
            const _Float16* __restrict__ bf =
                Bp + ((size_t)(tileN0 + t) * KB + kblk) * 512 + lane * 16;
            v8h b_lo = *reinterpret_cast<const v8h*>(bf);
            v8h b_hi = *reinterpret_cast<const v8h*>(bf + 8);
            v16h b = __builtin_shufflevector(b_lo, b_hi, 0, 1, 2, 3, 4, 5, 6, 7,
                                             8, 9, 10, 11, 12, 13, 14, 15);
            acc[t] = __builtin_amdgcn_wmma_f32_16x16x32_f16(
                false, a, false, b, (short)0, acc[t], false, false);
        }
    }

    #pragma unroll
    for (int t = 0; t < NT; ++t) {
        const int col = (tileN0 + t) * 16 + l;
        const float bv = add_bias ? bias[col] : 0.0f;
        #pragma unroll
        for (int v = 0; v < 8; ++v) {
            const int row = tileM * 16 + v + 8 * hi;
            C[(size_t)row * N + col] = acc[t][v] + bv;
        }
    }
}

// ---------------------------------------------------------------------------
// Aggregation: agg[n,f] = bias[f] + dinv[n]^2 * h[n,f]   (self-loop term)
// float4 vectorized; F multiple of 4.
// ---------------------------------------------------------------------------
__global__ void agg_init_kernel(const float* __restrict__ h,
                                const float* __restrict__ dinv,
                                const float* __restrict__ bias,
                                float* __restrict__ agg, int n, int F) {
    int idx = blockIdx.x * blockDim.x + threadIdx.x;   // quad index
    int F4 = F >> 2;
    if (idx >= n * F4) return;
    int node = idx / F4;
    int f4 = idx - node * F4;
    float d2 = dinv[node];
    d2 *= d2;
    float4 hv = reinterpret_cast<const float4*>(h)[idx];
    float4 bvv = reinterpret_cast<const float4*>(bias)[f4];
    float4 o;
    o.x = bvv.x + d2 * hv.x;
    o.y = bvv.y + d2 * hv.y;
    o.z = bvv.z + d2 * hv.z;
    o.w = bvv.w + d2 * hv.w;
    reinterpret_cast<float4*>(agg)[idx] = o;
}

// One wave per edge; lanes stride the feature dimension; fp32 atomic adds.
template <int F>
__global__ void edge_scatter_kernel(const int* __restrict__ ei, int E,
                                    const float* __restrict__ h,
                                    const float* __restrict__ dinv,
                                    float* __restrict__ agg) {
    int wid  = (blockIdx.x * blockDim.x + threadIdx.x) >> 5;
    int lane = threadIdx.x & 31;
    if (wid >= E) return;
    int src = ei[wid];
    int dst = ei[E + wid];
    float norm = dinv[src] * dinv[dst];
    const float* __restrict__ hs = h + (size_t)src * F;
    float* __restrict__ ad = agg + (size_t)dst * F;
    #pragma unroll
    for (int r = 0; r < F / 32; ++r) {
        int f = lane + 32 * r;
        atomicAdd(&ad[f], hs[f] * norm);
    }
}

// ---------------------------------------------------------------------------
// In-place per-row log-softmax over 16 classes.
// ---------------------------------------------------------------------------
__global__ void log_softmax16_kernel(float* __restrict__ out, int n) {
    int i = blockIdx.x * blockDim.x + threadIdx.x;
    if (i >= n) return;
    float* row = out + (size_t)i * 16;
    float v[16];
    float m = -INFINITY;
    #pragma unroll
    for (int c = 0; c < 16; ++c) { v[c] = row[c]; m = fmaxf(m, v[c]); }
    float s = 0.0f;
    #pragma unroll
    for (int c = 0; c < 16; ++c) s += expf(v[c] - m);
    float ls = logf(s);
    #pragma unroll
    for (int c = 0; c < 16; ++c) row[c] = v[c] - m - ls;
}

// ---------------------------------------------------------------------------
// Host-side orchestration
// ---------------------------------------------------------------------------
extern "C" void kernel_launch(void* const* d_in, const int* in_sizes, int n_in,
                              void* d_out, int out_size, void* d_ws, size_t ws_size,
                              hipStream_t stream) {
    const float* x   = (const float*)d_in[0];   // [N, 128]
    const int*   ei  = (const int*)  d_in[1];   // [2, E]
    const float* W1  = (const float*)d_in[2];   // [128, 128]
    const float* b1  = (const float*)d_in[3];   // [128]
    const float* W2  = (const float*)d_in[4];   // [128, 256]
    const float* b2  = (const float*)d_in[5];   // [256]
    const float* Wfc = (const float*)d_in[6];   // [256, 16]
    const float* bfc = (const float*)d_in[7];   // [16]
    float* out = (float*)d_out;                 // [N, 16]

    const int F1 = 128, F2 = 256, NC = 16;
    const int N = in_sizes[0] / F1;             // 50000 (multiple of 16)
    const int E = in_sizes[1] / 2;              // 800000

    // Workspace carve-up: fp32 arrays first, then f16 regions (all 16B aligned).
    float* wsf  = (float*)d_ws;
    float* dinv = wsf;                 wsf += N;
    float* XW1  = wsf;                 wsf += (size_t)N * F1;
    float* H1   = wsf;                 wsf += (size_t)N * F1;
    float* H1W2 = wsf;                 wsf += (size_t)N * F2;
    float* H2   = wsf;                 wsf += (size_t)N * F2;
    _Float16* wsh = (_Float16*)wsf;
    _Float16* Ah   = wsh;              wsh += (size_t)N * F2;  // reused: Xh/H1h/H2h
    _Float16* W1p  = wsh;              wsh += (size_t)F1 * F1;
    _Float16* W2p  = wsh;              wsh += (size_t)F1 * F2;
    _Float16* Wfcp = wsh;              wsh += (size_t)F2 * NC;

    const int TPB = 256;
    auto cdiv = [](long long a, long long b) { return (int)((a + b - 1) / b); };

    // --- degree / normalization ---
    init_deg_kernel<<<cdiv(N, TPB), TPB, 0, stream>>>(dinv, N);
    count_deg_kernel<<<cdiv(E, TPB), TPB, 0, stream>>>(ei, E, dinv);
    finalize_dinv_kernel<<<cdiv(N, TPB), TPB, 0, stream>>>(dinv, N);

    // --- one-time weight packing (tiny) ---
    pack_b_kernel<<<cdiv(F1 * F1, TPB), TPB, 0, stream>>>(W1, W1p, F1, F1);
    pack_b_kernel<<<cdiv(F1 * F2, TPB), TPB, 0, stream>>>(W2, W2p, F1, F2);
    pack_b_kernel<<<cdiv(F2 * NC, TPB), TPB, 0, stream>>>(Wfc, Wfcp, F2, NC);

    // --- layer 1: XW1 = x @ W1 ; H1 = Ahat(XW1) + b1 ---
    pack_a_kernel<<<cdiv((long long)N * F1 / 8, TPB), TPB, 0, stream>>>(
        x, Ah, N * F1 / 8, 0);
    {
        long long waves = (long long)(N / 16) * (F1 / 16 / 4);
        wmma_gemm_packed_kernel<4, 128><<<cdiv(waves * 32, TPB), TPB, 0, stream>>>(
            Ah, W1p, nullptr, XW1, N, F1, 0);
    }
    agg_init_kernel<<<cdiv((long long)N * F1 / 4, TPB), TPB, 0, stream>>>(
        XW1, dinv, b1, H1, N, F1);
    edge_scatter_kernel<128><<<cdiv((long long)E * 32, TPB), TPB, 0, stream>>>(
        ei, E, XW1, dinv, H1);

    // --- layer 2: H1W2 = relu(H1) @ W2 ; H2 = Ahat(H1W2) + b2 ---
    pack_a_kernel<<<cdiv((long long)N * F1 / 8, TPB), TPB, 0, stream>>>(
        H1, Ah, N * F1 / 8, 1);
    {
        long long waves = (long long)(N / 16) * (F2 / 16 / 4);
        wmma_gemm_packed_kernel<4, 128><<<cdiv(waves * 32, TPB), TPB, 0, stream>>>(
            Ah, W2p, nullptr, H1W2, N, F2, 0);
    }
    agg_init_kernel<<<cdiv((long long)N * F2 / 4, TPB), TPB, 0, stream>>>(
        H1W2, dinv, b2, H2, N, F2);
    edge_scatter_kernel<256><<<cdiv((long long)E * 32, TPB), TPB, 0, stream>>>(
        ei, E, H1W2, dinv, H2);

    // --- classifier: out = H2 @ Wfc + bfc ; log_softmax in place ---
    pack_a_kernel<<<cdiv((long long)N * F2 / 8, TPB), TPB, 0, stream>>>(
        H2, Ah, N * F2 / 8, 0);
    {
        long long waves = (long long)(N / 16) * (NC / 16);
        wmma_gemm_packed_kernel<1, 256><<<cdiv(waves * 32, TPB), TPB, 0, stream>>>(
            Ah, Wfcp, bfc, out, N, NC, 1);
    }
    log_softmax16_kernel<<<cdiv(N, TPB), TPB, 0, stream>>>(out, N);
}
